// FEAST_layer_73005854097931
// MI455X (gfx1250) — matrix operand order repaired and checked by reading, not compiled, not run on hardware
//
#include <hip/hip_runtime.h>
#include <hip/hip_fp16.h>

typedef __attribute__((ext_vector_type(16))) _Float16 v16h;
typedef __attribute__((ext_vector_type(8)))  float    v8f;

__device__ __forceinline__ float lrelu(float x) { return x >= 0.f ? x : 0.01f * x; }

// Deterministic float atomic max via signed/unsigned integer monotonicity trick.
// Works with -inf initialization.
__device__ __forceinline__ void atomicMaxF(float* addr, float val) {
    if (val >= 0.f) atomicMax((int*)addr, __float_as_int(val));
    else            atomicMin((unsigned int*)addr, (unsigned int)__float_as_int(val));
}

// ---------------------------------------------------------------------------
// K0: zero outputs, init maxes to -inf, denominators to 0
// ---------------------------------------------------------------------------
__global__ void k_init(float* __restrict__ out, float* __restrict__ m_o,
                       float* __restrict__ m_a, float* __restrict__ den_o,
                       float* __restrict__ den_a, int n64, int n2) {
    int i = blockIdx.x * blockDim.x + threadIdx.x;
    if (i < n64) out[i] = 0.f;
    if (i < n2) { m_o[i] = -INFINITY; m_a[i] = -INFINITY; den_o[i] = 0.f; den_a[i] = 0.f; }
}

// ---------------------------------------------------------------------------
// K1: WMMA node transforms.
//   mat 0: [th | lh]  = h  @ [w1 | w2]  + [b1 | b2]
//   mat 1: [tah| lah] = ah @ [wa1| wa2] + [ba1| ba2]
// One wave computes one 16-node x 64-col tile: 4 col-tiles x 2 K-steps of
// v_wmma_f32_16x16x32_f16 (f32 accumulate).  A-row index is clamped for loads
// (out-of-range rows are never stored), so there is a single branch-free load
// path; stores use a wave-uniform full-tile fast path.
// ---------------------------------------------------------------------------
__global__ __launch_bounds__(256) void k_transform(
    const float* __restrict__ h,  const float* __restrict__ ah,
    const float* __restrict__ w1, const float* __restrict__ b1,
    const float* __restrict__ wa1,const float* __restrict__ ba1,
    const float* __restrict__ w2, const float* __restrict__ b2,
    const float* __restrict__ wa2,const float* __restrict__ ba2,
    float* __restrict__ th, float* __restrict__ tah,
    float* __restrict__ lh, float* __restrict__ lah, int N)
{
    int wid  = (blockIdx.x * blockDim.x + threadIdx.x) >> 5;
    int lane = threadIdx.x & 31;
    int tiles = (N + 15) >> 4;
    if (wid >= 2 * tiles) return;          // wave-uniform: EXEC stays all-1s for WMMA
    int mat  = wid & 1;
    int tile = wid >> 1;

    const float* X  = mat ? ah  : h;
    const float* WT = mat ? wa1 : w1;      // 64x32 row-major (K x N)
    const float* WL = mat ? wa2 : w2;
    const float* BT = mat ? ba1 : b1;
    const float* BL = mat ? ba2 : b2;
    float* OT = mat ? tah : th;
    float* OL = mat ? lah : lh;

    int g  = lane >> 4;                    // half-wave group
    int lm = lane & 15;
    int nodeA = tile * 16 + lm;            // A-matrix row this lane supplies
    if (nodeA >= N) nodeA = N - 1;         // clamp: loads safe, stores guarded
    bool full = (tile * 16 + 16) <= N;     // wave-uniform

    // A layout (16-bit A 16x32, ISA 7.12.2): lanes 0-15 -> M=lane, K in
    // {g*8..g*8+7} for halves 0..7 and {16+g*8..} for halves 8..15.
    v16h a0, a1;
    {
        const float4* Xv = (const float4*)(X + (size_t)nodeA * 64);
        float4 c0a = Xv[g * 2],      c0b = Xv[g * 2 + 1];       // K = g*8 .. g*8+7
        float4 c1a = Xv[4 + g * 2],  c1b = Xv[5 + g * 2];       // K = 16+g*8 ..
        float4 c2a = Xv[8 + g * 2],  c2b = Xv[9 + g * 2];       // K = 32+g*8 ..
        float4 c3a = Xv[12 + g * 2], c3b = Xv[13 + g * 2];      // K = 48+g*8 ..
        a0[0]=(_Float16)c0a.x; a0[1]=(_Float16)c0a.y; a0[2]=(_Float16)c0a.z; a0[3]=(_Float16)c0a.w;
        a0[4]=(_Float16)c0b.x; a0[5]=(_Float16)c0b.y; a0[6]=(_Float16)c0b.z; a0[7]=(_Float16)c0b.w;
        a0[8]=(_Float16)c1a.x; a0[9]=(_Float16)c1a.y; a0[10]=(_Float16)c1a.z; a0[11]=(_Float16)c1a.w;
        a0[12]=(_Float16)c1b.x; a0[13]=(_Float16)c1b.y; a0[14]=(_Float16)c1b.z; a0[15]=(_Float16)c1b.w;
        a1[0]=(_Float16)c2a.x; a1[1]=(_Float16)c2a.y; a1[2]=(_Float16)c2a.z; a1[3]=(_Float16)c2a.w;
        a1[4]=(_Float16)c2b.x; a1[5]=(_Float16)c2b.y; a1[6]=(_Float16)c2b.z; a1[7]=(_Float16)c2b.w;
        a1[8]=(_Float16)c3a.x; a1[9]=(_Float16)c3a.y; a1[10]=(_Float16)c3a.z; a1[11]=(_Float16)c3a.w;
        a1[12]=(_Float16)c3b.x; a1[13]=(_Float16)c3b.y; a1[14]=(_Float16)c3b.z; a1[15]=(_Float16)c3b.w;
    }

    #pragma unroll
    for (int tc = 0; tc < 4; ++tc) {
        // tc is compile-time after unroll -> W/bias/output selection is scalar
        const float* W = (tc < 2) ? WT : WL;
        const float* B = (tc < 2) ? BT : BL;
        float*       O = (tc < 2) ? OT : OL;
        int col = (tc & 1) * 16 + lm;          // 0..31 within selected matrix
        // B layout (16-bit B 32x16): lane = N, half i holds B[g*8 + (i&7) + (i&8)*2][N]
        v16h b0, b1v;
        #pragma unroll
        for (int i = 0; i < 16; ++i) {
            int k = g * 8 + (i & 7) + (i & 8) * 2;
            b0[i]  = (_Float16)W[k * 32 + col];
            b1v[i] = (_Float16)W[(k + 32) * 32 + col];
        }
        v8f c = {};
        c = __builtin_amdgcn_wmma_f32_16x16x32_f16(false, a0, false, b0,  (short)0, c, false, false);
        c = __builtin_amdgcn_wmma_f32_16x16x32_f16(false, a1, false, b1v, (short)0, c, false, false);

        float bias = B[col];
        // C layout: VGPR r -> row M = r + 8*g, col = lane&15
        if (full) {
            #pragma unroll
            for (int r = 0; r < 8; ++r) {
                int node = tile * 16 + r + 8 * g;
                O[(size_t)node * 32 + col] = c[r] + bias;
            }
        } else {
            #pragma unroll
            for (int r = 0; r < 8; ++r) {
                int node = tile * 16 + r + 8 * g;
                if (node < N) O[(size_t)node * 32 + col] = c[r] + bias;
            }
        }
    }
}

// ---------------------------------------------------------------------------
// K2: per-node score partials.  Every edge score is a dot of a concatenation,
// so it splits into src-part + dst-part, computed once per node instead of
// once per edge (16x less work: N=50K vs E=800K).
//   rs = <h, wr[0:64]> + <ah, wr[64:128]>      rd = <h,wr[128:192]>+<ah,wr[192:]>
//   per head: [ap1 an1 bp1 bn1 | ap2 an2 bp2 bn2]  (a=th, b=tah; p=wp, n=wn;
//   1 = first 16 rows of the 32x1 weight, 2 = last 16)
// ---------------------------------------------------------------------------
__global__ void k_nodescal(const float* __restrict__ h,  const float* __restrict__ ah,
                           const float* __restrict__ th, const float* __restrict__ tah,
                           const float* __restrict__ wp, const float* __restrict__ wn,
                           const float* __restrict__ wr,
                           float* __restrict__ rs, float* __restrict__ rd,
                           float* __restrict__ ascal, int N)
{
    int n = blockIdx.x * blockDim.x + threadIdx.x;
    if (n >= N) return;
    const float4* hv4  = (const float4*)(h  + (size_t)n * 64);
    const float4* av4  = (const float4*)(ah + (size_t)n * 64);
    const float4* wr0  = (const float4*)(wr);
    const float4* wr1  = (const float4*)(wr + 64);
    const float4* wr2  = (const float4*)(wr + 128);
    const float4* wr3  = (const float4*)(wr + 192);
    float s0 = 0.f, s1 = 0.f;
    #pragma unroll
    for (int q = 0; q < 16; ++q) {
        float4 hv = hv4[q], av = av4[q];
        float4 a = wr0[q], b = wr1[q], c = wr2[q], d = wr3[q];
        s0 += hv.x*a.x + hv.y*a.y + hv.z*a.z + hv.w*a.w
            + av.x*b.x + av.y*b.y + av.z*b.z + av.w*b.w;
        s1 += hv.x*c.x + hv.y*c.y + hv.z*c.z + hv.w*c.w
            + av.x*d.x + av.y*d.y + av.z*d.z + av.w*d.w;
    }
    rs[n] = s0; rd[n] = s1;

    #pragma unroll
    for (int hh = 0; hh < 2; ++hh) {
        const float4* t4  = (const float4*)(th  + (size_t)n * 32 + hh * 16);
        const float4* ta4 = (const float4*)(tah + (size_t)n * 32 + hh * 16);
        float ap1=0,an1=0,bp1=0,bn1=0,ap2=0,an2=0,bp2=0,bn2=0;
        #pragma unroll
        for (int q = 0; q < 4; ++q) {
            float4 t  = t4[q];
            float4 ta = ta4[q];
            float4 p1 = ((const float4*)wp)[q];
            float4 p2 = ((const float4*)(wp + 16))[q];
            float4 q1 = ((const float4*)wn)[q];
            float4 q2 = ((const float4*)(wn + 16))[q];
            ap1 += t.x*p1.x + t.y*p1.y + t.z*p1.z + t.w*p1.w;
            ap2 += t.x*p2.x + t.y*p2.y + t.z*p2.z + t.w*p2.w;
            an1 += t.x*q1.x + t.y*q1.y + t.z*q1.z + t.w*q1.w;
            an2 += t.x*q2.x + t.y*q2.y + t.z*q2.z + t.w*q2.w;
            bp1 += ta.x*p1.x + ta.y*p1.y + ta.z*p1.z + ta.w*p1.w;
            bp2 += ta.x*p2.x + ta.y*p2.y + ta.z*p2.z + ta.w*p2.w;
            bn1 += ta.x*q1.x + ta.y*q1.y + ta.z*q1.z + ta.w*q1.w;
            bn2 += ta.x*q2.x + ta.y*q2.y + ta.z*q2.z + ta.w*q2.w;
        }
        float4* o = (float4*)(ascal + ((size_t)n * 2 + hh) * 8);
        o[0] = make_float4(ap1, an1, bp1, bn1);   // src-side float4
        o[1] = make_float4(ap2, an2, bp2, bn2);   // dst-side float4
    }
}

__device__ __forceinline__ void edge_scores(const float* __restrict__ ascal,
                                            int s, int d, int hh, bool pos,
                                            float BP, float BN,
                                            float& so, float& sa) {
    float4 sv = *(const float4*)(ascal + ((size_t)s * 2 + hh) * 8);      // ap1 an1 bp1 bn1
    float4 dv = *(const float4*)(ascal + ((size_t)d * 2 + hh) * 8 + 4);  // ap2 an2 bp2 bn2
    if (pos) { so = lrelu(sv.x + dv.x + BP); sa = lrelu(sv.z + dv.z + BP); }
    else     { so = lrelu(sv.w + dv.y + BN); sa = lrelu(sv.y + dv.w + BN); }
}

// ---------------------------------------------------------------------------
// K3: segment max of edge scores into m_o/m_a
// ---------------------------------------------------------------------------
__global__ void k_edgemax(const int* __restrict__ src, const int* __restrict__ dst,
                          const float* __restrict__ rs, const float* __restrict__ rd,
                          const float* __restrict__ ascal,
                          const float* __restrict__ bp, const float* __restrict__ bn,
                          const float* __restrict__ br,
                          float* __restrict__ m_o, float* __restrict__ m_a, int E)
{
    int e = blockIdx.x * blockDim.x + threadIdx.x;
    if (e >= E) return;
    int s = src[e], d = dst[e];
    bool pos = (rs[s] + rd[d] + br[0]) >= 0.f;
    float BP = bp[0], BN = bn[0];
    #pragma unroll
    for (int hh = 0; hh < 2; ++hh) {
        float so, sa;
        edge_scores(ascal, s, d, hh, pos, BP, BN, so, sa);
        atomicMaxF(&m_o[d * 2 + hh], so);
        atomicMaxF(&m_a[d * 2 + hh], sa);
    }
}

// ---------------------------------------------------------------------------
// K4: exp + accumulate denominator and weighted features (into d_out directly)
// ---------------------------------------------------------------------------
__global__ void k_edgeacc(const int* __restrict__ src, const int* __restrict__ dst,
                          const float* __restrict__ rs, const float* __restrict__ rd,
                          const float* __restrict__ ascal,
                          const float* __restrict__ th, const float* __restrict__ tah,
                          const float* __restrict__ bp, const float* __restrict__ bn,
                          const float* __restrict__ br,
                          const float* __restrict__ m_o, const float* __restrict__ m_a,
                          float* __restrict__ den_o, float* __restrict__ den_a,
                          float* __restrict__ out, float* __restrict__ aout, int E)
{
    int e = blockIdx.x * blockDim.x + threadIdx.x;
    if (e >= E) return;
    int s = src[e], d = dst[e];
    bool pos = (rs[s] + rd[d] + br[0]) >= 0.f;
    float BP = bp[0], BN = bn[0];
    const float* fo = pos ? th  : tah;   // feat_o = pos ? src_h  : src_ah (transformed)
    const float* fa = pos ? tah : th;    // feat_a = pos ? src_ah : src_h
    #pragma unroll
    for (int hh = 0; hh < 2; ++hh) {
        float so, sa;
        edge_scores(ascal, s, d, hh, pos, BP, BN, so, sa);
        float eo = __expf(so - m_o[d * 2 + hh]);
        float ea = __expf(sa - m_a[d * 2 + hh]);
        atomicAdd(&den_o[d * 2 + hh], eo);
        atomicAdd(&den_a[d * 2 + hh], ea);
        // 16 contiguous floats per head -> 4x global_load_b128 each
        const float4* fo4 = (const float4*)(fo + (size_t)s * 32 + hh * 16);
        const float4* fa4 = (const float4*)(fa + (size_t)s * 32 + hh * 16);
        #pragma unroll
        for (int q = 0; q < 4; ++q) {
            float4 vo = fo4[q];
            float4 va = fa4[q];
            int ob = d * 32 + hh * 16 + q * 4;
            atomicAdd(&out[ob + 0], eo * vo.x);
            atomicAdd(&out[ob + 1], eo * vo.y);
            atomicAdd(&out[ob + 2], eo * vo.z);
            atomicAdd(&out[ob + 3], eo * vo.w);
            atomicAdd(&aout[ob + 0], ea * va.x);
            atomicAdd(&aout[ob + 1], ea * va.y);
            atomicAdd(&aout[ob + 2], ea * va.z);
            atomicAdd(&aout[ob + 3], ea * va.w);
        }
    }
}

// ---------------------------------------------------------------------------
// K5: normalize and add linear path
// ---------------------------------------------------------------------------
__global__ void k_final(float* __restrict__ out, float* __restrict__ aout,
                        const float* __restrict__ lh, const float* __restrict__ lah,
                        const float* __restrict__ den_o, const float* __restrict__ den_a,
                        int N)
{
    int idx = blockIdx.x * blockDim.x + threadIdx.x;
    if (idx >= N * 32) return;
    int n  = idx >> 5;
    int hh = (idx >> 4) & 1;
    float d_o = fmaxf(den_o[n * 2 + hh], 1e-16f);
    float d_a = fmaxf(den_a[n * 2 + hh], 1e-16f);
    out[idx]  = out[idx]  / d_o + lh[idx];
    aout[idx] = aout[idx] / d_a + lah[idx];
}

extern "C" void kernel_launch(void* const* d_in, const int* in_sizes, int n_in,
                              void* d_out, int out_size, void* d_ws, size_t ws_size,
                              hipStream_t stream) {
    const float* h   = (const float*)d_in[0];
    const float* ah  = (const float*)d_in[1];
    const float* w1  = (const float*)d_in[2];
    const float* b1  = (const float*)d_in[3];
    const float* wa1 = (const float*)d_in[4];
    const float* ba1 = (const float*)d_in[5];
    const float* w2  = (const float*)d_in[6];
    const float* b2  = (const float*)d_in[7];
    const float* wa2 = (const float*)d_in[8];
    const float* ba2 = (const float*)d_in[9];
    const float* wp  = (const float*)d_in[10];
    const float* bp  = (const float*)d_in[11];
    const float* wn  = (const float*)d_in[12];
    const float* bn  = (const float*)d_in[13];
    const float* wr  = (const float*)d_in[14];
    const float* br  = (const float*)d_in[15];
    const int*   src = (const int*)d_in[16];
    const int*   dst = (const int*)d_in[17];

    const int N = in_sizes[0] / 64;
    const int E = in_sizes[16];

    // workspace layout (floats): 4 node-transform buffers, rs/rd, 16 score
    // partials per node, maxes + denominators.  Total = N*154 floats (~31 MB).
    float* ws    = (float*)d_ws;
    float* th    = ws;
    float* tah   = th  + (size_t)N * 32;
    float* lh    = tah + (size_t)N * 32;
    float* lah   = lh  + (size_t)N * 32;
    float* rs    = lah + (size_t)N * 32;
    float* rd    = rs  + N;
    float* ascal = rd  + N;
    float* m_o   = ascal + (size_t)N * 16;
    float* m_a   = m_o + (size_t)N * 2;
    float* den_o = m_a + (size_t)N * 2;
    float* den_a = den_o + (size_t)N * 2;

    float* out  = (float*)d_out;
    float* aout = out + (size_t)N * 32;

    int n64 = N * 64, n2 = N * 2;
    k_init<<<(n64 + 255) / 256, 256, 0, stream>>>(out, m_o, m_a, den_o, den_a, n64, n2);

    int tiles = (N + 15) / 16;
    int threads = 2 * tiles * 32;   // one wave per (tile, matrix)
    k_transform<<<(threads + 255) / 256, 256, 0, stream>>>(
        h, ah, w1, b1, wa1, ba1, w2, b2, wa2, ba2, th, tah, lh, lah, N);

    k_nodescal<<<(N + 255) / 256, 256, 0, stream>>>(
        h, ah, th, tah, wp, wn, wr, rs, rd, ascal, N);

    k_edgemax<<<(E + 255) / 256, 256, 0, stream>>>(
        src, dst, rs, rd, ascal, bp, bn, br, m_o, m_a, E);

    k_edgeacc<<<(E + 255) / 256, 256, 0, stream>>>(
        src, dst, rs, rd, ascal, th, tah, bp, bn, br, m_o, m_a,
        den_o, den_a, out, aout, E);

    k_final<<<(N * 32 + 255) / 256, 256, 0, stream>>>(
        out, aout, lh, lah, den_o, den_a, N);
}
